// RelativePositionTransformerEncoderLayer_7413113553636
// MI455X (gfx1250) — compile-verified, matrix operand order checked by reading, not run on hardware
//
#include <hip/hip_runtime.h>
#include <hip/hip_bf16.h>

typedef _Float16 f16_t;
typedef __attribute__((ext_vector_type(4)))  _Float16 v4h;
typedef __attribute__((ext_vector_type(16))) _Float16 v16h;
typedef __attribute__((ext_vector_type(8)))  float    v8f;

#define D_MODEL 768
#define LSEQ    1024
#define NHEAD   12
#define DK      64
#define MROWS   2048   // B*L
#define LDT     40     // LDS tile row stride in halves (16B-aligned, bank-spread)

// ---------------------------------------------------------------------------
// Fragment helpers (CDNA5 WMMA 16x16x32 f16 layouts, ISA 7.12.2)
// A (16x32, MxK): lane l -> m = l&15, khalf = l>>4; two contiguous 8-half runs:
//   K = khalf*8 + i (i<8) and K = 16 + khalf*8 + i.
// B (32x16, KxN) loaded from a transposed view (row = n, col = k), same pattern.
// C/D (16x16 f32): lane l -> n = l&15; VGPR r -> M = (l>=16?8:0)+r.
// ---------------------------------------------------------------------------

__device__ __forceinline__ v16h frag_from_lds(const f16_t* row, int khalf) {
  v16h f;
  const f16_t* p0 = row + khalf * 8;
  const f16_t* p1 = p0 + 16;
#pragma unroll
  for (int i = 0; i < 8; ++i) { f[i] = p0[i]; f[i + 8] = p1[i]; }
  return f;
}

__device__ __forceinline__ v16h frag_from_f32(const float* row, int khalf, float scale) {
  v16h f;
  const float* p0 = row + khalf * 8;
  const float* p1 = p0 + 16;
#pragma unroll
  for (int i = 0; i < 8; ++i) {
    f[i]     = (_Float16)(p0[i] * scale);
    f[i + 8] = (_Float16)(p1[i] * scale);
  }
  return f;
}

__device__ __forceinline__ v4h pack4(float a, float b, float c, float d) {
  auto lo = __builtin_amdgcn_cvt_pkrtz(a, b);   // v_cvt_pk_rtz_f16_f32
  auto hi = __builtin_amdgcn_cvt_pkrtz(c, d);
  v4h p;
  p[0] = (_Float16)lo[0]; p[1] = (_Float16)lo[1];
  p[2] = (_Float16)hi[0]; p[3] = (_Float16)hi[1];
  return p;
}

// ---------------------------------------------------------------------------
// Generic GEMM:  C[M,N] = A[M,K] @ W[K,N] + bias[N]  (optional ReLU)
// fp32 in/out, f16 WMMA compute. Tile 128(M) x 128(N) x 32(K), 256 threads.
// Waves 4(M) x 2(N); each wave owns 32x64 (2x4 WMMA tiles, 8 WMMA / K-step).
// Register-buffered pipeline: fetch(next tile -> regs) overlaps WMMA; commit
// (regs -> LDS, packed b64) at the top of each iteration; double LDS buffers.
// ---------------------------------------------------------------------------
template <int RELU>
__global__ void __launch_bounds__(256)
gemm_wmma(const float* __restrict__ A, const float* __restrict__ W,
          const float* __restrict__ bias, float* __restrict__ C,
          int M, int N, int K)
{
  __shared__ __align__(16) f16_t As[2][128][LDT];   // row-major A tile
  __shared__ __align__(16) f16_t Bs[2][128][LDT];   // W^T tile: [n][k]

  const int tid   = threadIdx.x;
  const int lane  = tid & 31;
  const int w     = tid >> 5;        // 0..7
  const int wm    = w >> 1;          // 0..3 (M blocks of 32)
  const int wn    = w & 1;           // 0..1 (N blocks of 64)
  const int bm    = blockIdx.y * 128;
  const int bn    = blockIdx.x * 128;
  const int mr    = lane & 15;
  const int khalf = lane >> 4;

  v8f acc[2][4] = {};
  float4 areg[4];
  float  breg[4][4];

  auto fetch = [&](int k0) {
#pragma unroll
    for (int it = 0; it < 4; ++it) {
      const int qi = it * 256 + tid;           // 0..1023 quad index
      const int r = qi >> 3, c = (qi & 7) * 4; // A: contiguous float4
      areg[it] = *(const float4*)(A + (size_t)(bm + r) * K + k0 + c);
      const int n = qi & 127, kq = (qi >> 7) * 4;  // B: 4 k-strided dwords
      const float* wp = W + (size_t)(k0 + kq) * N + bn + n;
#pragma unroll
      for (int j = 0; j < 4; ++j) breg[it][j] = wp[(size_t)j * N];
    }
  };
  auto commit = [&](int buf) {
#pragma unroll
    for (int it = 0; it < 4; ++it) {
      const int qi = it * 256 + tid;
      const int r = qi >> 3, c = (qi & 7) * 4;
      *(v4h*)&As[buf][r][c] = pack4(areg[it].x, areg[it].y, areg[it].z, areg[it].w);
      const int n = qi & 127, kq = (qi >> 7) * 4;
      *(v4h*)&Bs[buf][n][kq] = pack4(breg[it][0], breg[it][1], breg[it][2], breg[it][3]);
    }
  };

  fetch(0);
  int buf = 0;
  for (int k0 = 0; k0 < K; k0 += 32) {
    commit(buf);
    __syncthreads();
    if (k0 + 32 < K) fetch(k0 + 32);   // global loads in flight during WMMAs

    v16h af[2], bf[4];
#pragma unroll
    for (int i = 0; i < 2; ++i)
      af[i] = frag_from_lds(&As[buf][wm * 32 + i * 16 + mr][0], khalf);
#pragma unroll
    for (int j = 0; j < 4; ++j)
      bf[j] = frag_from_lds(&Bs[buf][wn * 64 + j * 16 + mr][0], khalf);
#pragma unroll
    for (int i = 0; i < 2; ++i)
#pragma unroll
      for (int j = 0; j < 4; ++j)
        acc[i][j] = __builtin_amdgcn_wmma_f32_16x16x32_f16(
            false, af[i], false, bf[j], (short)0, acc[i][j], false, false);
    buf ^= 1;
  }

  const int nlane = lane & 15;
  const int mbase = (lane >> 4) * 8;
#pragma unroll
  for (int i = 0; i < 2; ++i)
#pragma unroll
    for (int j = 0; j < 4; ++j) {
      int col = bn + wn * 64 + j * 16 + nlane;
      float bv = bias[col];
#pragma unroll
      for (int r = 0; r < 8; ++r) {
        int row = bm + wm * 32 + i * 16 + mbase + r;
        float v = acc[i][j][r] + bv;
        if (RELU) v = fmaxf(v, 0.0f);
        C[(size_t)row * N + col] = v;
      }
    }
}

// ---------------------------------------------------------------------------
// Fused attention: per (b, h, 16-query tile). 4 waves / 128 threads.
// scores = (Q/8) @ K^T + biasMLP(rel) -> softmax -> @V
// Dynamic LDS: S[16][1024] f32 + cached kx/ky + MLP params + reduction pad.
// ---------------------------------------------------------------------------
__global__ void __launch_bounds__(128)
attn_kernel(const float* __restrict__ Q, const float* __restrict__ Kb,
            const float* __restrict__ V, const float* __restrict__ sx,
            const float* __restrict__ sy, const float* __restrict__ P1,
            const float* __restrict__ pb1, const float* __restrict__ P2,
            const float* __restrict__ pb2, float* __restrict__ O)
{
  extern __shared__ float smem[];
  float* S    = smem;                 // 16*1024
  float* kx   = S + 16 * LSEQ;        // 1024
  float* ky   = kx + LSEQ;            // 1024
  float* p1s  = ky + LSEQ;            // 64  (P1 row-major: [0][j], [1][j])
  float* b1s  = p1s + 64;             // 32
  float* p2c  = b1s + 32;             // 32  (P2[:, h])
  float* qxv  = p2c + 32;             // 16
  float* qyv  = qxv + 16;             // 16
  float* red  = qyv + 16;             // 16*8

  const int tid  = threadIdx.x;
  const int lane = tid & 31;
  const int w    = tid >> 5;          // 0..3
  const int qb   = blockIdx.x * 16;
  const int h    = blockIdx.y;
  const int b    = blockIdx.z;
  const size_t rowbase = (size_t)b * LSEQ;
  const int mr    = lane & 15;
  const int khalf = lane >> 4;

  for (int i = tid; i < LSEQ; i += 128) {
    kx[i] = sx[rowbase + i];
    ky[i] = sy[rowbase + i];
  }
  for (int i = tid; i < 64; i += 128) p1s[i] = P1[i];
  if (tid < 32) { b1s[tid] = pb1[tid]; p2c[tid] = P2[tid * NHEAD + h]; }
  if (tid < 16) { qxv[tid] = sx[rowbase + qb + tid]; qyv[tid] = sy[rowbase + qb + tid]; }
  __syncthreads();
  const float pb2h = pb2[h];

  // Q fragments in registers (scale 1/sqrt(64) folded in)
  const float* qrow = Q + (rowbase + qb + mr) * D_MODEL + h * DK;
  v16h qf[2];
#pragma unroll
  for (int c = 0; c < 2; ++c) qf[c] = frag_from_f32(qrow + c * 32, khalf, 0.125f);

  // ---- scores + bias MLP ----
  for (int kt = w; kt < LSEQ / 16; kt += 4) {
    const int key = kt * 16 + mr;
    const float* krow = Kb + (rowbase + key) * D_MODEL + h * DK;
    v8f acc = {};
#pragma unroll
    for (int c = 0; c < 2; ++c) {
      v16h bf = frag_from_f32(krow + c * 32, khalf, 1.0f);
      acc = __builtin_amdgcn_wmma_f32_16x16x32_f16(
          false, qf[c], false, bf, (short)0, acc, false, false);
    }
    const float kxv = kx[key], kyv = ky[key];
#pragma unroll
    for (int r = 0; r < 8; ++r) {
      int m = khalf * 8 + r;
      float rx = qxv[m] - kxv;
      float ry = qyv[m] - kyv;
      rx = fminf(fmaxf(rx, -1000.0f), 1000.0f) * 0.001f;
      ry = fminf(fmaxf(ry, -1000.0f), 1000.0f) * 0.001f;
      float bsum = pb2h;
      for (int j = 0; j < 32; ++j) {
        float hj = fmaxf(rx * p1s[j] + ry * p1s[32 + j] + b1s[j], 0.0f);
        bsum += hj * p2c[j];
      }
      S[m * LSEQ + key] = acc[r] + bsum;
    }
  }
  __syncthreads();

  // ---- softmax over each of 16 rows (8 threads per row) ----
  {
    const int r = tid >> 3;
    const int s = tid & 7;
    float mx = -1e30f;
    for (int c = s * 128; c < s * 128 + 128; ++c) mx = fmaxf(mx, S[r * LSEQ + c]);
    red[r * 8 + s] = mx;
    __syncthreads();
    if (s == 0) {
      float m2 = red[r * 8];
      for (int t = 1; t < 8; ++t) m2 = fmaxf(m2, red[r * 8 + t]);
      red[r * 8] = m2;
    }
    __syncthreads();
    const float rm = red[r * 8];
    float sum = 0.0f;
    for (int c = s * 128; c < s * 128 + 128; ++c) {
      float e = __expf(S[r * LSEQ + c] - rm);
      S[r * LSEQ + c] = e;
      sum += e;
    }
    __syncthreads();
    red[r * 8 + s] = sum;
    __syncthreads();
    if (s == 0) {
      float t = 0.0f;
      for (int u = 0; u < 8; ++u) t += red[r * 8 + u];
      red[r * 8] = 1.0f / t;
    }
    __syncthreads();
    const float inv = red[r * 8];
    for (int c = s * 128; c < s * 128 + 128; ++c) S[r * LSEQ + c] *= inv;
  }
  __syncthreads();

  // ---- O = P @ V  (wave w owns dk columns [w*16, w*16+16)) ----
  v8f oacc = {};
  const int ncol = lane & 15;
  for (int kb = 0; kb < LSEQ / 32; ++kb) {
    const float* srow = &S[mr * LSEQ + kb * 32];
    v16h af = frag_from_f32(srow, khalf, 1.0f);
    // B fragment from V (strided: column h*64 + w*16 + ncol, rows = keys)
    const float* vb = V + (rowbase + kb * 32 + khalf * 8) * D_MODEL + h * DK + w * 16 + ncol;
    v16h bf;
#pragma unroll
    for (int i = 0; i < 8; ++i) {
      bf[i]     = (_Float16)vb[(size_t)i * D_MODEL];
      bf[i + 8] = (_Float16)vb[(size_t)(i + 16) * D_MODEL];
    }
    oacc = __builtin_amdgcn_wmma_f32_16x16x32_f16(
        false, af, false, bf, (short)0, oacc, false, false);
  }
  const int mb2 = khalf * 8;
#pragma unroll
  for (int r = 0; r < 8; ++r)
    O[(rowbase + qb + mb2 + r) * D_MODEL + h * DK + w * 16 + ncol] = oacc[r];
}

// ---------------------------------------------------------------------------
// out = LayerNorm(x + y) * g + be   (one block per row of 768)
// ---------------------------------------------------------------------------
__global__ void __launch_bounds__(256)
add_ln(const float* __restrict__ x, const float* __restrict__ y,
       const float* __restrict__ g, const float* __restrict__ be,
       float* __restrict__ out)
{
  __shared__ float rs[256];
  __shared__ float rq[256];
  const int row = blockIdx.x;
  const int tid = threadIdx.x;
  const float* xr = x + (size_t)row * D_MODEL;
  const float* yr = y + (size_t)row * D_MODEL;
  float v[3], s = 0.0f, s2 = 0.0f;
#pragma unroll
  for (int i = 0; i < 3; ++i) {
    float t = xr[tid + i * 256] + yr[tid + i * 256];
    v[i] = t; s += t; s2 += t * t;
  }
  rs[tid] = s; rq[tid] = s2;
  __syncthreads();
  for (int off = 128; off > 0; off >>= 1) {
    if (tid < off) { rs[tid] += rs[tid + off]; rq[tid] += rq[tid + off]; }
    __syncthreads();
  }
  const float mean = rs[0] * (1.0f / D_MODEL);
  const float var  = rq[0] * (1.0f / D_MODEL) - mean * mean;
  const float inv  = rsqrtf(var + 1e-5f);
#pragma unroll
  for (int i = 0; i < 3; ++i) {
    int c = tid + i * 256;
    out[(size_t)row * D_MODEL + c] = (v[i] - mean) * inv * g[c] + be[c];
  }
}

// ---------------------------------------------------------------------------
extern "C" void kernel_launch(void* const* d_in, const int* in_sizes, int n_in,
                              void* d_out, int out_size, void* d_ws, size_t ws_size,
                              hipStream_t stream)
{
  const float* src = (const float*)d_in[0];
  const float* sx  = (const float*)d_in[1];
  const float* sy  = (const float*)d_in[2];
  const float* Wq  = (const float*)d_in[3];  const float* bq  = (const float*)d_in[4];
  const float* Wk  = (const float*)d_in[5];  const float* bk  = (const float*)d_in[6];
  const float* Wv  = (const float*)d_in[7];  const float* bv  = (const float*)d_in[8];
  const float* Wo  = (const float*)d_in[9];  const float* bo  = (const float*)d_in[10];
  const float* P1  = (const float*)d_in[11]; const float* pb1 = (const float*)d_in[12];
  const float* P2  = (const float*)d_in[13]; const float* pb2 = (const float*)d_in[14];
  const float* W1  = (const float*)d_in[15]; const float* b1  = (const float*)d_in[16];
  const float* W2  = (const float*)d_in[17]; const float* b2  = (const float*)d_in[18];
  const float* g1  = (const float*)d_in[19]; const float* be1 = (const float*)d_in[20];
  const float* g2  = (const float*)d_in[21]; const float* be2 = (const float*)d_in[22];

  float* ws = (float*)d_ws;
  const size_t NT = (size_t)MROWS * D_MODEL;   // 1,572,864 floats
  float* Qb = ws;            // [0, NT)
  float* Kf = ws + NT;       // [NT, 2NT)
  float* Vb = ws + 2 * NT;   // [2NT, 3NT)
  float* Ob = ws + 3 * NT;   // [3NT, 4NT)
  float* X1 = ws + 4 * NT;   // [4NT, 5NT)
  float* Ff = ws + 5 * NT;   // [5NT, 6NT)
  float* S2 = Qb;            // out-proj result reuses Q region
  float* Hh = ws;            // FFN hidden (2048x3072) reuses [0, 4NT)

  dim3 g768(D_MODEL / 128, MROWS / 128);       // (6, 16)
  dim3 gffn(3072 / 128, MROWS / 128);          // (24, 16)

  // QKV projections
  gemm_wmma<0><<<g768, 256, 0, stream>>>(src, Wq, bq, Qb, MROWS, D_MODEL, D_MODEL);
  gemm_wmma<0><<<g768, 256, 0, stream>>>(src, Wk, bk, Kf, MROWS, D_MODEL, D_MODEL);
  gemm_wmma<0><<<g768, 256, 0, stream>>>(src, Wv, bv, Vb, MROWS, D_MODEL, D_MODEL);

  // fused attention (dynamic LDS)
  size_t smem = (size_t)(16 * LSEQ + 2 * LSEQ + 64 + 32 + 32 + 16 + 16 + 128) * sizeof(float);
  attn_kernel<<<dim3(LSEQ / 16, NHEAD, 2), 128, smem, stream>>>(
      Qb, Kf, Vb, sx, sy, P1, pb1, P2, pb2, Ob);

  // output projection + residual LN 1
  gemm_wmma<0><<<g768, 256, 0, stream>>>(Ob, Wo, bo, S2, MROWS, D_MODEL, D_MODEL);
  add_ln<<<MROWS, 256, 0, stream>>>(src, S2, g1, be1, X1);

  // FFN
  gemm_wmma<1><<<gffn, 256, 0, stream>>>(X1, W1, b1, Hh, MROWS, 3072, D_MODEL);
  gemm_wmma<0><<<g768, 256, 0, stream>>>(Hh, W2, b2, Ff, MROWS, D_MODEL, 3072);

  // residual LN 2 -> output
  add_ln<<<MROWS, 256, 0, stream>>>(X1, Ff, g2, be2, (float*)d_out);
}
